// AbstractGenerativeUpsample_63780264346313
// MI455X (gfx1250) — matrix operand order repaired and checked by reading, not compiled, not run on hardware
//
#include <hip/hip_runtime.h>
#include <hip/hip_bf16.h>

typedef __attribute__((ext_vector_type(16))) __bf16 v16bf;
typedef __attribute__((ext_vector_type(8)))  float  v8f;

#define N_PARENT 50000
#define C_IN     256
#define C_OUT    256
#define FANOUT   8
#define N_TGT    200000
#define N_UP     (N_PARENT * FANOUT)      // 400000
#define ROWS_WG  16
#define KSTEPS   8                        // 256 / 32
#define FRAG_US  (32 * 16)                // ushorts per fragment (1 KB)

__device__ __forceinline__ unsigned short f2bf(float f) {
    unsigned u = __builtin_bit_cast(unsigned, f);
    u += 0x7FFFu + ((u >> 16) & 1u);      // round-to-nearest-even
    return (unsigned short)(u >> 16);
}

// ---------------- target init / scatter ----------------
__global__ void init_target_kernel(float* __restrict__ tgt, int n) {
    int i = blockIdx.x * blockDim.x + threadIdx.x;
    if (i < n) tgt[i] = 0.0f;
}

__global__ void scatter_target_kernel(const int* __restrict__ idx,
                                      float* __restrict__ tgt, int n, int nup) {
    int i = blockIdx.x * blockDim.x + threadIdx.x;
    if (i < n) {
        int r = idx[i];
        if ((unsigned)r < (unsigned)nup) tgt[r] = 1.0f;   // idempotent race is fine
    }
}

// ---------------- W_up -> bf16 fragment-order swizzle ----------------
// wsB flat index = ((((k*8 + kk)*16 + nt)*32 + lane)*16 + j)
// element = W[k][kk*32 + Klocal][nt*16 + (lane&15)], Klocal = (lane<16?0:8)+(j<8?j:j+8)
__global__ void swizzle_w_kernel(const float* __restrict__ W,
                                 unsigned short* __restrict__ wsB) {
    int t = blockIdx.x * blockDim.x + threadIdx.x;      // 0 .. 524287
    if (t >= FANOUT * KSTEPS * 16 * 32 * 16) return;
    int j    =  t        & 15;
    int lane = (t >> 4)  & 31;
    int nt   = (t >> 9)  & 15;
    int kk   = (t >> 13) & 7;
    int k    =  t >> 16;
    int n      = nt * 16 + (lane & 15);
    int Klocal = ((lane < 16) ? 0 : 8) + ((j < 8) ? j : j + 8);
    float f = W[((size_t)k * C_IN + (kk * 32 + Klocal)) * C_OUT + n];
    wsB[t] = f2bf(f);
}

// ---------------- fused upsample GEMM + classify + prune ----------------
__global__ __launch_bounds__(256) void gen_upsample_kernel(
    const float* __restrict__ fea,
    const unsigned short* __restrict__ wsB,
    const float* __restrict__ b_up,
    const float* __restrict__ W_cls,
    const float* __restrict__ b_cls,
    const float* __restrict__ tgt,        // 0/1 floats (output region 3)
    float* __restrict__ out_fea,
    float* __restrict__ out_exist)
{
    __shared__ __align__(32) unsigned short Alds[KSTEPS * FRAG_US]; // 8 KB
    __shared__ float exist_lds[ROWS_WG];
    __shared__ float keep_lds[ROWS_WG];

    const int tid  = threadIdx.x;
    const int w    = tid >> 5;            // wave 0..7
    const int lane = tid & 31;
    const int rowBase = blockIdx.x * ROWS_WG;

    // ---- stage A tile (16 rows x 256) into LDS in A-fragment order ----
    // A-frag: lane l holds row m=l&15; kbase=(l<16?0:8); elem j -> K=kbase+(j<8?j:j+8)
    #pragma unroll
    for (int pp = 0; pp < 8; ++pp) {
        int p   = tid * 8 + pp;           // 0..2047 (pairs of bf16)
        int kk  = p >> 8;
        int rem = p & 255;
        int l   = rem >> 3;
        int j   = (rem & 7) * 2;
        int m     = l & 15;
        int kbase = (l < 16) ? 0 : 8;
        int K0    = kbase + ((j < 8) ? j : j + 8);
        const float* src = fea + (size_t)(rowBase + m) * C_IN + kk * 32 + K0;
        unsigned lo = f2bf(src[0]);
        unsigned hi = f2bf(src[1]);
        *(unsigned*)(&Alds[(kk * 32 + l) * 16 + j]) = lo | (hi << 16);
    }
    __syncthreads();

    // ---- pull A fragments into registers (shared by all 8 waves) ----
    v16bf a[KSTEPS];
    #pragma unroll
    for (int kk = 0; kk < KSTEPS; ++kk)
        a[kk] = *(const v16bf*)(&Alds[(kk * 32 + lane) * 16]);

    const int nc = lane & 15;
    const int n0 = w * 32 + nc;           // this wave's two N-tiles
    const int n1 = n0 + 16;
    const float bu0  = b_up[n0];
    const float bu1  = b_up[n1];
    const float wc0  = W_cls[n0];
    const float wc1  = W_cls[n1];
    const float bcls = b_cls[0];
    const int mhalf  = (lane >> 4) << 3;  // rows 0..7 vs 8..15 per C/D layout

    for (int k = 0; k < FANOUT; ++k) {
        v8f acc0 = {};
        v8f acc1 = {};
        const unsigned short* Bk = wsB + (size_t)k * (KSTEPS * 16 * FRAG_US);
        #pragma unroll
        for (int kk = 0; kk < KSTEPS; ++kk) {
            const unsigned short* Bf = Bk + (size_t)(kk * 16 + w * 2) * FRAG_US;
            v16bf b0 = *(const v16bf*)(Bf + lane * 16);
            v16bf b1 = *(const v16bf*)(Bf + (32 + lane) * 16);
            acc0 = __builtin_amdgcn_wmma_f32_16x16x32_bf16(
                       false, a[kk], false, b0, (short)0, acc0, false, false);
            acc1 = __builtin_amdgcn_wmma_f32_16x16x32_bf16(
                       false, a[kk], false, b1, (short)0, acc1, false, false);
        }

        // add upsample bias (per output column)
        #pragma unroll
        for (int v = 0; v < 8; ++v) { acc0[v] += bu0; acc1[v] += bu1; }

        // ---- classifier dot: exist[m] = sum_n fea_up[m][n] * W_cls[n] + b_cls ----
        __syncthreads();                          // exist_lds reuse across k
        if (tid < ROWS_WG) exist_lds[tid] = bcls;
        __syncthreads();
        #pragma unroll
        for (int v = 0; v < 8; ++v) {
            float p = acc0[v] * wc0 + acc1[v] * wc1;
            p += __shfl_xor(p, 1, 16);
            p += __shfl_xor(p, 2, 16);
            p += __shfl_xor(p, 4, 16);
            p += __shfl_xor(p, 8, 16);
            if ((lane & 15) == 0) atomicAdd(&exist_lds[mhalf + v], p);
        }
        __syncthreads();

        // ---- keep = (exist > 0) | target ; emit exist ----
        if (tid < ROWS_WG) {
            size_t r = (size_t)(rowBase + tid) * FANOUT + k;
            float e = exist_lds[tid];
            out_exist[r] = e;
            keep_lds[tid] = (e > 0.0f || tgt[r] > 0.5f) ? 1.0f : 0.0f;
        }
        __syncthreads();

        // ---- masked store of fea_pruned ----
        #pragma unroll
        for (int v = 0; v < 8; ++v) {
            int m = mhalf + v;
            float kf = keep_lds[m];
            size_t r = (size_t)(rowBase + m) * FANOUT + k;
            out_fea[r * C_OUT + n0] = acc0[v] * kf;
            out_fea[r * C_OUT + n1] = acc1[v] * kf;
        }
    }
}

extern "C" void kernel_launch(void* const* d_in, const int* in_sizes, int n_in,
                              void* d_out, int out_size, void* d_ws, size_t ws_size,
                              hipStream_t stream) {
    const float* fea   = (const float*)d_in[0];   // [50000,256]
    const float* W_up  = (const float*)d_in[1];   // [8,256,256]
    const float* b_up  = (const float*)d_in[2];   // [256]
    const float* W_cls = (const float*)d_in[3];   // [256,1]
    const float* b_cls = (const float*)d_in[4];   // [1]
    const int*   tidx  = (const int*)d_in[5];     // [200000]

    float* out_fea   = (float*)d_out;                         // 400000*256
    float* out_exist = out_fea + (size_t)N_UP * C_OUT;        // 400000
    float* out_tgt   = out_exist + N_UP;                      // 400000
    unsigned short* wsB = (unsigned short*)d_ws;              // 1 MB swizzled bf16 W

    init_target_kernel<<<(N_UP + 255) / 256, 256, 0, stream>>>(out_tgt, N_UP);
    scatter_target_kernel<<<(N_TGT + 255) / 256, 256, 0, stream>>>(tidx, out_tgt, N_TGT, N_UP);
    swizzle_w_kernel<<<(FANOUT * C_IN * C_OUT) / 256, 256, 0, stream>>>(W_up, wsB);
    gen_upsample_kernel<<<N_PARENT / ROWS_WG, 256, 0, stream>>>(
        fea, wsB, b_up, W_cls, b_cls, out_tgt, out_fea, out_exist);
}
